// IsoCA_5437428597211
// MI455X (gfx1250) — matrix operand order repaired and checked by reading, not compiled, not run on hardware
//
#include <hip/hip_runtime.h>

// ---------------------------------------------------------------------------
// Fused NCA step for gfx1250 (MI455X): conv features in f32 VALU, perceptron
// layers on v_wmma_f32_16x16x32_bf16, single pass over HBM (~105 MB).
// Layer loops interleave the wave's two pixel tiles: shared A fragments,
// independent accumulator chains (no WMMA->WMMA hazard NOPs), VGPRs < 256.
// ---------------------------------------------------------------------------

typedef __attribute__((ext_vector_type(16))) __bf16        v16bf;
typedef __attribute__((ext_vector_type(8)))  float         v8f;
typedef __attribute__((ext_vector_type(4)))  unsigned int  v4u;

union Frag {
    v16bf        v;
    v4u          q[2];
    unsigned int u[8];
};

__device__ __forceinline__ unsigned int f2bf(float f) {
    unsigned int u = __float_as_uint(f);
    return (u + 0x7FFFu + ((u >> 16) & 1u)) >> 16;   // round-to-nearest-even
}
__device__ __forceinline__ unsigned int pack2bf(float lo, float hi) {
    return f2bf(lo) | (f2bf(hi) << 16);
}

#define TB_H 8
#define TB_W 16
#define NPIX 128
#define PK 72     // perc panel row stride (halves): 144 B, 16B aligned, bank-spread
#define WK 72     // w1 panel row stride (halves)
#define HK 136    // hidden / w2 panel row stride (halves): 272 B

__global__ void __launch_bounds__(128)
nca_fused_kernel(const float* __restrict__ x,
                 const float* __restrict__ rnd,
                 const float* __restrict__ w1w,
                 const float* __restrict__ w1b,
                 const float* __restrict__ w2w,
                 float* __restrict__ out)
{
    __shared__ __align__(16) float          xs[12][10][20];       // haloed tile
    __shared__ __align__(16) unsigned short perc_s[NPIX * PK];    // [pixel][K=64]
    __shared__ __align__(16) unsigned short hid_s[NPIX * HK];     // [pixel][128]
    __shared__ __align__(16) unsigned short w1_s[128 * WK];       // [o][K=64]
    __shared__ __align__(16) unsigned short w2_s[16 * HK];        // [o16][128]
    __shared__ float bias_s[128];
    __shared__ float mlds[NPIX];                                  // mask*life

    const int tid  = threadIdx.x;
    const int lane = tid & 31;
    const int wid  = tid >> 5;
    const int b    = blockIdx.z;
    const int row0 = blockIdx.y * TB_H;
    const int col0 = blockIdx.x * TB_W;

    const float* xb = x + (long)b * 12 * 65536;

    // ---- Phase 0/1: cooperative loads --------------------------------------
    for (int i = tid; i < 12 * 10 * 18; i += 128) {
        int c   = i / 180;
        int rmd = i - c * 180;
        int hr  = rmd / 18;
        int hc  = rmd - hr * 18;
        int gr  = (row0 + hr - 1) & 255;
        int gc  = (col0 + hc - 1) & 255;
        xs[c][hr][hc] = xb[(c << 16) + (gr << 8) + gc];
    }
    {   // w1: 128x36 -> bf16, K zero-padded to 64 (one row per thread)
        unsigned int* wrow = (unsigned int*)&w1_s[tid * WK];
        const float*  src  = w1w + tid * 36;
        #pragma unroll
        for (int k2 = 0; k2 < 18; ++k2) wrow[k2] = pack2bf(src[2 * k2], src[2 * k2 + 1]);
        #pragma unroll
        for (int k2 = 18; k2 < 32; ++k2) wrow[k2] = 0u;
        bias_s[tid] = w1b[tid];
    }
    for (int i = tid; i < 16 * 128; i += 128) {   // w2: 9x128, M zero-padded to 16
        int o = i >> 7;
        int k = i & 127;
        w2_s[o * HK + k] = (o < 9) ? (unsigned short)f2bf(w2w[o * 128 + k])
                                   : (unsigned short)0;
    }
    __syncthreads();

    // ---- Phase 2: per-pixel conv features -> bf16 perception panel ---------
    const int pr = tid >> 4;   // pixel row in tile (0..7)
    const int pc = tid & 15;   // pixel col in tile (0..15)
    {
        unsigned int* prow = (unsigned int*)&perc_s[tid * PK];
        float g01[2];
        #pragma unroll
        for (int c2 = 0; c2 < 12; c2 += 2) {
            #pragma unroll
            for (int cc = 0; cc < 2; ++cc) {
                int c = c2 + cc;
                float v00 = xs[c][pr][pc],     v01 = xs[c][pr][pc + 1],     v02 = xs[c][pr][pc + 2];
                float v10 = xs[c][pr + 1][pc], v11 = xs[c][pr + 1][pc + 1], v12 = xs[c][pr + 1][pc + 2];
                float v20 = xs[c][pr + 2][pc], v21 = xs[c][pr + 2][pc + 1], v22 = xs[c][pr + 2][pc + 2];
                float gx  = (v02 - v00) + 2.f * (v12 - v10) + (v22 - v20);
                float gy  = (v20 - v00) + 2.f * (v21 - v01) + (v22 - v02);
                float lap = v00 + v02 + v20 + v22 + 2.f * (v01 + v10 + v12 + v21) - 12.f * v11;
                float gn  = __builtin_sqrtf(gx * gx + gy * gy + 1e-8f);
                prow[c]   = pack2bf(v11, lap);              // channels 2c, 2c+1
                g01[cc]   = gn;
            }
            prow[12 + (c2 >> 1)] = pack2bf(g01[0], g01[1]); // channels 24+c
        }
        #pragma unroll
        for (int k2 = 18; k2 < 32; ++k2) prow[k2] = 0u;     // zero-pad K 36..63
    }
    {   // update mask * life mask (max-pool ch3, -inf edge padding)
        int   gr   = row0 + pr, gc = col0 + pc;
        float mask = floorf(rnd[((long)b << 16) + (gr << 8) + gc] + 0.5f);
        float pm   = -__builtin_inff();
        #pragma unroll
        for (int dy = -1; dy <= 1; ++dy) {
            int yy = gr + dy;
            #pragma unroll
            for (int dx = -1; dx <= 1; ++dx) {
                int xx = gc + dx;
                bool inb = (yy >= 0) & (yy < 256) & (xx >= 0) & (xx < 256);
                float v = xs[3][pr + 1 + dy][pc + 1 + dx];
                pm = inb ? fmaxf(pm, v) : pm;
            }
        }
        mlds[tid] = mask * ((pm > 0.1f) ? 1.f : 0.f);
    }
    {   // gene channels 9..11 pass through unchanged
        long obase = ((long)b * 12) * 65536 + (long)(row0 + pr) * 256 + (col0 + pc);
        #pragma unroll
        for (int g = 9; g < 12; ++g)
            out[obase + (long)g * 65536] = xs[g][pr + 1][pc + 1];
    }
    __syncthreads();

    // ---- WMMA fragment geometry (ISA 7.12.2, wave32) -----------------------
    const int n    = lane & 15;        // B/C column, A row within 16
    const int half = lane >> 4;
    const int kbA  = half * 16;        // A K-base
    const int kbB  = half * 16;        // B K-base
    const int hi8  = half * 8;         // C/D M offset

    const int t0 = 2 * wid;            // wave owns pixel tiles t0, t0+1
    const int pA = t0 * 16 + n;        // == wave's own lanes' pixels
    const int pB = pA + 16;

    // ---- Layer 1: 36->128; A shared across both tiles, 2 indep acc chains --
    {
        Frag b0A, b1A, b0B, b1B;
        {
            const v4u* bqA = (const v4u*)&perc_s[pA * PK + kbB];
            const v4u* bqB = (const v4u*)&perc_s[pB * PK + kbB];
            b0A.q[0] = bqA[0]; b0A.q[1] = bqA[1];   // K =  kbB .. kbB+15
            b1A.q[0] = bqA[4]; b1A.q[1] = bqA[5];   // K = 32+kbB .. 32+kbB+15
            b0B.q[0] = bqB[0]; b0B.q[1] = bqB[1];
            b1B.q[0] = bqB[4]; b1B.q[1] = bqB[5];
        }
        #pragma unroll 2
        for (int m = 0; m < 8; ++m) {
            const v4u* aq = (const v4u*)&w1_s[(m * 16 + n) * WK + kbA];
            Frag a0, a1;
            a0.q[0] = aq[0]; a0.q[1] = aq[1];       // K = kbA..kbA+15
            a1.q[0] = aq[4]; a1.q[1] = aq[5];       // K0 = 32
            v8f accA = {}, accB = {};
            accA = __builtin_amdgcn_wmma_f32_16x16x32_bf16(false, a0.v, false, b0A.v,
                                                           (short)0, accA, false, false);
            accB = __builtin_amdgcn_wmma_f32_16x16x32_bf16(false, a0.v, false, b0B.v,
                                                           (short)0, accB, false, false);
            accA = __builtin_amdgcn_wmma_f32_16x16x32_bf16(false, a1.v, false, b1A.v,
                                                           (short)0, accA, false, false);
            accB = __builtin_amdgcn_wmma_f32_16x16x32_bf16(false, a1.v, false, b1B.v,
                                                           (short)0, accB, false, false);
            // epilogue: bias + leaky relu, repack bf16 into hidden panel
            #pragma unroll
            for (int tt = 0; tt < 2; ++tt) {
                v8f acc = tt ? accB : accA;
                int p   = tt ? pB : pA;
                v4u hv;
                #pragma unroll
                for (int r2 = 0; r2 < 4; ++r2) {
                    float y0 = acc[2 * r2]     + bias_s[m * 16 + hi8 + 2 * r2];
                    float y1 = acc[2 * r2 + 1] + bias_s[m * 16 + hi8 + 2 * r2 + 1];
                    y0 = (y0 > 0.f) ? y0 : 0.01f * y0;
                    y1 = (y1 > 0.f) ? y1 : 0.01f * y1;
                    hv[r2] = pack2bf(y0, y1);
                }
                *(v4u*)&hid_s[p * HK + m * 16 + hi8] = hv;   // 16B ds_store_b128
            }
        }
    }
    __syncthreads();

    // ---- Layer 2: 128->9(padded 16); shared A, 2 indep acc chains ----------
    {
        const v4u* aq  = (const v4u*)&w2_s[n * HK + kbA];
        const v4u* bqA = (const v4u*)&hid_s[pA * HK + kbB];
        const v4u* bqB = (const v4u*)&hid_s[pB * HK + kbB];
        v8f accA = {}, accB = {};
        #pragma unroll
        for (int kk = 0; kk < 4; ++kk) {
            Frag a, bA, bB;
            a.q[0]  = aq[kk * 4];  a.q[1]  = aq[kk * 4 + 1];
            bA.q[0] = bqA[kk * 4]; bA.q[1] = bqA[kk * 4 + 1];
            bB.q[0] = bqB[kk * 4]; bB.q[1] = bqB[kk * 4 + 1];
            accA = __builtin_amdgcn_wmma_f32_16x16x32_bf16(false, a.v, false, bA.v,
                                                           (short)0, accA, false, false);
            accB = __builtin_amdgcn_wmma_f32_16x16x32_bf16(false, a.v, false, bB.v,
                                                           (short)0, accB, false, false);
        }
        // masked residual update, coalesced per channel plane
        #pragma unroll
        for (int tt = 0; tt < 2; ++tt) {
            v8f   acc = tt ? accB : accA;
            int   p   = tt ? pB : pA;
            int   t   = t0 + tt;
            float ml  = mlds[p];
            long  obase = ((long)b * 12) * 65536 + (long)(row0 + t) * 256 + (col0 + n);
            #pragma unroll
            for (int r = 0; r < 8; ++r) {
                int o = hi8 + r;                       // output channel (M)
                if (o < 9) {
                    float xin = xs[o][t + 1][n + 1];
                    out[obase + (long)o * 65536] = xin + acc[r] * ml;
                }
            }
        }
    }
}

extern "C" void kernel_launch(void* const* d_in, const int* in_sizes, int n_in,
                              void* d_out, int out_size, void* d_ws, size_t ws_size,
                              hipStream_t stream) {
    const float* x   = (const float*)d_in[0];
    const float* rnd = (const float*)d_in[1];
    const float* w1w = (const float*)d_in[2];
    const float* w1b = (const float*)d_in[3];
    const float* w2w = (const float*)d_in[4];
    float* out = (float*)d_out;

    dim3 grid(256 / TB_W, 256 / TB_H, 16);   // 16 x 32 x 16 blocks
    nca_fused_kernel<<<grid, 128, 0, stream>>>(x, rnd, w1w, w1b, w2w, out);
}
